// GraphAttentionNetwork_77747497992411
// MI455X (gfx1250) — compile-verified, matrix operand order checked by reading, not run on hardware
//
#include <hip/hip_runtime.h>
#include <hip/hip_fp16.h>
#include <math.h>

typedef _Float16 v16h __attribute__((ext_vector_type(16)));
typedef _Float16 v8h  __attribute__((ext_vector_type(8)));
typedef float    v8f  __attribute__((ext_vector_type(8)));

#define NEG_SLOPE 0.2f

__device__ __forceinline__ float lrelu(float x) { return x > 0.f ? x : NEG_SLOPE * x; }

// float atomic max via sign-split int trick (valid when dest initialized to -inf)
__device__ __forceinline__ void atomicMaxF(float* addr, float val) {
    if (val >= 0.f) atomicMax((int*)addr, __float_as_int(val));
    else            atomicMin((unsigned int*)addr, __float_as_uint(val));
}

// ---------------------------------------------------------------- utilities
__global__ void k_fill(float* __restrict__ p, float v, int n) {
    int i = blockIdx.x * blockDim.x + threadIdx.x;
    if (i < n) p[i] = v;
}

__global__ void k_f32_to_f16(const float* __restrict__ in, _Float16* __restrict__ out, int n) {
    int i = blockIdx.x * blockDim.x + threadIdx.x;
    if (i < n) out[i] = (_Float16)in[i];
}

// Pack row-major f32 W[K][Ncols] into the CDNA5 16-bit B-fragment layout:
// packed[((nt*KS + ks)*32 + lane)*16 + e]  where for lane group (lane<16 ? k0=0 : k0=8):
//   e in 0..7  -> K = ks*32 + k0 + e          (VGPR 0..3, two halves each)
//   e in 8..15 -> K = ks*32 + 16 + k0 + (e-8) (VGPR 4..7)
//   N = nt*16 + (lane & 15)
__global__ void k_pack_w(const float* __restrict__ W, _Float16* __restrict__ packed,
                         int K, int Ncols) {
    int idx = blockIdx.x * blockDim.x + threadIdx.x;
    int KS = K >> 5;
    int total = (Ncols >> 4) * KS * 32 * 16;
    if (idx >= total) return;
    int e    = idx & 15;
    int lane = (idx >> 4) & 31;
    int rest = idx >> 9;             // nt*KS + ks
    int nt = rest / KS, ks = rest % KS;
    int col = nt * 16 + (lane & 15);
    int k0  = (lane & 16) ? 8 : 0;
    int k   = ks * 32 + ((e < 8) ? (k0 + e) : (16 + k0 + (e - 8)));
    packed[idx] = (_Float16)W[(size_t)k * Ncols + col];
}

// ---------------------------------------------------------------- WMMA GEMM
// C[M,128] = A[M,K](f16 row-major) x B(K,128)(pre-packed fragments), f32 accum.
// One 16x16 tile per wave; 8 waves per block cover the full 128-wide output row-tile.
__global__ void __launch_bounds__(256)
k_wmma_gemm(const _Float16* __restrict__ A, const _Float16* __restrict__ Bp,
            float* __restrict__ C, int K, int lda, int M) {
    const int lane = threadIdx.x & 31;
    const int w    = threadIdx.x >> 5;          // N-tile 0..7
    const int tm   = blockIdx.x;                // M-tile
    int row = tm * 16 + (lane & 15);
    if (row >= M) row = M - 1;                  // clamp (M is a multiple of 16 here)
    const int k0 = (lane & 16) ? 8 : 0;
    const int KS = K >> 5;

    const _Float16* __restrict__ Arow = A + (size_t)row * lda;
    const _Float16* __restrict__ Bw   = Bp + ((size_t)w * KS * 32 + lane) * 16;

    v8f acc = {};
    for (int ks = 0; ks < KS; ++ks) {
        const int kk = ks << 5;
        __builtin_prefetch(Arow + kk + 64, 0, 1);     // global_prefetch_b8
        v8h lo = *(const v8h*)(Arow + kk + k0);       // K = kk+k0 .. +7
        v8h hi = *(const v8h*)(Arow + kk + 16 + k0);  // K = kk+16+k0 .. +7
        v16h a;
#pragma unroll
        for (int i = 0; i < 8; ++i) { a[i] = lo[i]; a[i + 8] = hi[i]; }
        v16h b = *(const v16h*)(Bw + (size_t)ks * 32 * 16);
        acc = __builtin_amdgcn_wmma_f32_16x16x32_f16(
            /*neg_a=*/false, a, /*neg_b=*/false, b,
            /*c_mod=*/(short)0, acc, /*reuse_a=*/false, /*reuse_b=*/false);
    }
    // C/D layout: VGPR j -> M = tm*16 + j + (lane<16?0:8), N = w*16 + (lane&15)
    const int col   = w * 16 + (lane & 15);
    const int rbase = tm * 16 + ((lane & 16) ? 8 : 0);
#pragma unroll
    for (int j = 0; j < 8; ++j) {
        int r = rbase + j;
        if (r < M) C[(size_t)r * 128 + col] = acc[j];
    }
}

// ------------------------------------------------------- attention pipeline
// h layout [N][8 heads][16]; i = node*8 + head
__global__ void k_att(const float* __restrict__ h, const float* __restrict__ att_s,
                      const float* __restrict__ att_d, float* __restrict__ as_,
                      float* __restrict__ ad_, int n) {
    int i = blockIdx.x * blockDim.x + threadIdx.x;
    if (i >= n) return;
    int hh = i & 7;
    const float* hp = h + (size_t)i * 16;
    float s = 0.f, d = 0.f;
#pragma unroll
    for (int j = 0; j < 16; ++j) {
        float v = hp[j];
        s += v * att_s[hh * 16 + j];
        d += v * att_d[hh * 16 + j];
    }
    as_[i] = s; ad_[i] = d;
}

// segment max over edges (thread per edge*head); self-loops appended as edges [E, E+N)
__global__ void k_edge_max(const int* __restrict__ esrc, const int* __restrict__ edst,
                           const float* __restrict__ as_, const float* __restrict__ ad_,
                           float* __restrict__ m, int E, int ETot, int H) {
    int i = blockIdx.x * blockDim.x + threadIdx.x;
    if (i >= ETot * H) return;
    int hh = i % H, e = i / H;
    int s = (e < E) ? esrc[e] : (e - E);
    int d = (e < E) ? edst[e] : (e - E);
    atomicMaxF(&m[(size_t)d * H + hh], lrelu(as_[(size_t)s * H + hh] + ad_[(size_t)d * H + hh]));
}

// exp-sum + weighted numerator: 32 lanes per edge, 4 features per lane (HID=128, H=8)
__global__ void k_edge_agg(const int* __restrict__ esrc, const int* __restrict__ edst,
                           const float* __restrict__ hlin, const float* __restrict__ as_,
                           const float* __restrict__ ad_, const float* __restrict__ m,
                           float* __restrict__ den, float* __restrict__ num,
                           int E, int ETot) {
    int gid = blockIdx.x * blockDim.x + threadIdx.x;
    int e = gid >> 5, lane = gid & 31;
    if (e >= ETot) return;
    int s = (e < E) ? esrc[e] : (e - E);
    int d = (e < E) ? edst[e] : (e - E);
    float ex[4];
#pragma unroll
    for (int i = 0; i < 4; ++i) {
        int f  = lane + 32 * i;
        int hh = f >> 4;
        float mm = m[(size_t)d * 8 + hh];
        if (!isfinite(mm)) mm = 0.f;
        float ee = lrelu(as_[(size_t)s * 8 + hh] + ad_[(size_t)d * 8 + hh]);
        ex[i] = __expf(ee - mm);
    }
    if (lane < 8) {
        float mm = m[(size_t)d * 8 + lane];
        if (!isfinite(mm)) mm = 0.f;
        float ee = lrelu(as_[(size_t)s * 8 + lane] + ad_[(size_t)d * 8 + lane]);
        atomicAdd(&den[(size_t)d * 8 + lane], __expf(ee - mm));
    }
#pragma unroll
    for (int i = 0; i < 4; ++i) {
        int f = lane + 32 * i;
        atomicAdd(&num[(size_t)d * 128 + f], hlin[(size_t)s * 128 + f] * ex[i]);
    }
}

// normalize + bias (+ELU) and emit f16 copy for the next layer's GEMM
__global__ void k_norm_elu(const float* __restrict__ num, const float* __restrict__ den,
                           const float* __restrict__ bias, _Float16* __restrict__ out16,
                           int n, int doElu) {
    int i = blockIdx.x * blockDim.x + threadIdx.x;
    if (i >= n) return;
    int f = i & 127, hh = f >> 4, node = i >> 7;
    float v = num[i] / (den[(size_t)node * 8 + hh] + 1e-16f) + bias[f];
    if (doElu) v = v > 0.f ? v : (__expf(v) - 1.f);
    out16[i] = (_Float16)v;
}

// -------------------------------------------------------------- layer 3 (H=1,C=6)
__global__ void k_gemm3(const _Float16* __restrict__ h16, const float* __restrict__ W3,
                        float* __restrict__ h3, int N) {
    int n = blockIdx.x * blockDim.x + threadIdx.x;
    if (n >= N) return;
    float acc[6] = {0.f, 0.f, 0.f, 0.f, 0.f, 0.f};
    const _Float16* hp = h16 + (size_t)n * 128;
    for (int k = 0; k < 128; ++k) {
        float hv = (float)hp[k];
#pragma unroll
        for (int c = 0; c < 6; ++c) acc[c] += hv * W3[k * 6 + c];
    }
#pragma unroll
    for (int c = 0; c < 6; ++c) h3[(size_t)n * 6 + c] = acc[c];
}

__global__ void k_att3(const float* __restrict__ h3, const float* __restrict__ ws,
                       const float* __restrict__ wd, float* __restrict__ as_,
                       float* __restrict__ ad_, int N) {
    int i = blockIdx.x * blockDim.x + threadIdx.x;
    if (i >= N) return;
    float s = 0.f, d = 0.f;
#pragma unroll
    for (int c = 0; c < 6; ++c) {
        float v = h3[(size_t)i * 6 + c];
        s += v * ws[c]; d += v * wd[c];
    }
    as_[i] = s; ad_[i] = d;
}

__global__ void k_edge_agg3(const int* __restrict__ esrc, const int* __restrict__ edst,
                            const float* __restrict__ h3, const float* __restrict__ as_,
                            const float* __restrict__ ad_, const float* __restrict__ m,
                            float* __restrict__ den, float* __restrict__ num,
                            int E, int ETot) {
    int e = blockIdx.x * blockDim.x + threadIdx.x;
    if (e >= ETot) return;
    int s = (e < E) ? esrc[e] : (e - E);
    int d = (e < E) ? edst[e] : (e - E);
    float mm = m[d];
    if (!isfinite(mm)) mm = 0.f;
    float ex = __expf(lrelu(as_[s] + ad_[d]) - mm);
    atomicAdd(&den[d], ex);
#pragma unroll
    for (int c = 0; c < 6; ++c)
        atomicAdd(&num[(size_t)d * 6 + c], h3[(size_t)s * 6 + c] * ex);
}

__global__ void k_final(const float* __restrict__ num, const float* __restrict__ den,
                        const float* __restrict__ b3, float* __restrict__ out, int n) {
    int i = blockIdx.x * blockDim.x + threadIdx.x;
    if (i >= n) return;
    int c = i % 6, node = i / 6;
    out[i] = num[i] / (den[node] + 1e-16f) + b3[c];
}

// ---------------------------------------------------------------- launcher
extern "C" void kernel_launch(void* const* d_in, const int* in_sizes, int n_in,
                              void* d_out, int out_size, void* d_ws, size_t ws_size,
                              hipStream_t stream) {
    const float* x   = (const float*)d_in[0];
    const int*   ei  = (const int*)d_in[1];
    const float* W1  = (const float*)d_in[2];
    const float* b1  = (const float*)d_in[3];
    const float* at_s1 = (const float*)d_in[4];
    const float* at_d1 = (const float*)d_in[5];
    const float* W2  = (const float*)d_in[6];
    const float* b2  = (const float*)d_in[7];
    const float* at_s2 = (const float*)d_in[8];
    const float* at_d2 = (const float*)d_in[9];
    const float* W3  = (const float*)d_in[10];
    const float* b3  = (const float*)d_in[11];
    const float* at_s3 = (const float*)d_in[12];
    const float* at_d3 = (const float*)d_in[13];
    (void)n_in;

    const int F_IN = 256, HID = 128, H = 8, C = 6;
    const int N = in_sizes[0] / F_IN;
    const int E = in_sizes[1] / 2;
    const int ETot = E + N;
    const int* esrc = ei;
    const int* edst = ei + E;

    // workspace carve-up (256B aligned)
    char* ws = (char*)d_ws;
    size_t off = 0;
    auto alloc = [&](size_t bytes) -> char* {
        char* p = ws + off;
        off += (bytes + 255) & ~(size_t)255;
        return p;
    };
    _Float16* x16  = (_Float16*)alloc((size_t)N * F_IN * 2);  // aliased as `num` after GEMM1
    _Float16* h16  = (_Float16*)alloc((size_t)N * HID * 2);
    float* hlin    = (float*)alloc((size_t)N * HID * 4);
    float* a_s     = (float*)alloc((size_t)N * H * 4);
    float* a_d     = (float*)alloc((size_t)N * H * 4);
    float* m       = (float*)alloc((size_t)N * H * 4);
    float* den     = (float*)alloc((size_t)N * H * 4);
    _Float16* pW1  = (_Float16*)alloc((size_t)(F_IN / 32) * 8 * 32 * 16 * 2);
    _Float16* pW2  = (_Float16*)alloc((size_t)(HID / 32) * 8 * 32 * 16 * 2);
    float* h3      = (float*)alloc((size_t)N * C * 4);
    float* num3    = (float*)alloc((size_t)N * C * 4);
    float* a_s3    = (float*)alloc((size_t)N * 4);
    float* a_d3    = (float*)alloc((size_t)N * 4);
    float* m3      = (float*)alloc((size_t)N * 4);
    float* den3    = (float*)alloc((size_t)N * 4);
    float* num = (float*)x16;           // N*128*4 bytes == N*256*2 bytes: exact alias
    if (off > ws_size) return;          // insufficient scratch (should not happen)

    const int TB = 256;
    auto blk = [&](long long n) { return (unsigned)((n + TB - 1) / TB); };
    const float NEG_INF = -INFINITY;
    const int mtiles = (N + 15) / 16;

    // stage inputs
    k_f32_to_f16<<<blk((long long)N * F_IN), TB, 0, stream>>>(x, x16, N * F_IN);
    k_pack_w<<<blk((HID / 16) * (F_IN / 32) * 512), TB, 0, stream>>>(W1, pW1, F_IN, HID);
    k_pack_w<<<blk((HID / 16) * (HID / 32) * 512), TB, 0, stream>>>(W2, pW2, HID, HID);

    // ---- layers 1 & 2 ----
    for (int layer = 0; layer < 2; ++layer) {
        const _Float16* Ain  = (layer == 0) ? x16 : h16;
        const _Float16* Bpk  = (layer == 0) ? pW1 : pW2;
        const int K          = (layer == 0) ? F_IN : HID;
        const float* bias    = (layer == 0) ? b1 : b2;
        const float* atts    = (layer == 0) ? at_s1 : at_s2;
        const float* attd    = (layer == 0) ? at_d1 : at_d2;

        k_wmma_gemm<<<mtiles, TB, 0, stream>>>(Ain, Bpk, hlin, K, K, N);
        k_att<<<blk((long long)N * H), TB, 0, stream>>>(hlin, atts, attd, a_s, a_d, N * H);
        k_fill<<<blk((long long)N * H), TB, 0, stream>>>(m, NEG_INF, N * H);
        k_fill<<<blk((long long)N * H), TB, 0, stream>>>(den, 0.f, N * H);
        k_fill<<<blk((long long)N * HID), TB, 0, stream>>>(num, 0.f, N * HID);
        k_edge_max<<<blk((long long)ETot * H), TB, 0, stream>>>(esrc, edst, a_s, a_d, m, E, ETot, H);
        k_edge_agg<<<blk((long long)ETot * 32), TB, 0, stream>>>(esrc, edst, hlin, a_s, a_d, m,
                                                                 den, num, E, ETot);
        k_norm_elu<<<blk((long long)N * HID), TB, 0, stream>>>(num, den, bias, h16, N * HID, 1);
    }

    // ---- layer 3 (H=1, C=6, mean over 1 head == identity) ----
    k_gemm3<<<blk(N), TB, 0, stream>>>(h16, W3, h3, N);
    k_att3<<<blk(N), TB, 0, stream>>>(h3, at_s3, at_d3, a_s3, a_d3, N);
    k_fill<<<blk(N), TB, 0, stream>>>(m3, NEG_INF, N);
    k_fill<<<blk(N), TB, 0, stream>>>(den3, 0.f, N);
    k_fill<<<blk((long long)N * C), TB, 0, stream>>>(num3, 0.f, N * C);
    k_edge_max<<<blk(ETot), TB, 0, stream>>>(esrc, edst, a_s3, a_d3, m3, E, ETot, 1);
    k_edge_agg3<<<blk(ETot), TB, 0, stream>>>(esrc, edst, h3, a_s3, a_d3, m3, den3, num3, E, ETot);
    k_final<<<blk((long long)N * C), TB, 0, stream>>>(num3, den3, b3, (float*)d_out, N * C);
    (void)out_size;
}
